// BSNet_27384711479947
// MI455X (gfx1250) — compile-verified, matrix operand order checked by reading, not run on hardware
//
#include <hip/hip_runtime.h>

// ---------------------------------------------------------------------------
// BSNet on MI455X (gfx1250): all matmuls via v_wmma_f32_16x16x32_bf16;
// GEMM weight tiles staged with global_load_async_to_lds_b128 (ASYNCcnt).
// Workspace budget ~305 MB (documented in kernel_launch).
// ---------------------------------------------------------------------------

#define TOKS 14336           // total tokens in every stage (28*512 == 2048*7)
#define EPS_GN 1.1920929e-07f

typedef __attribute__((ext_vector_type(8)))  float  v8f;
typedef __attribute__((ext_vector_type(8)))  __bf16 v8bf;
typedef __attribute__((ext_vector_type(16))) __bf16 v16bf;

union fragu { v16bf v; v8bf h[2]; };

__device__ __forceinline__ __bf16 f2b(float f) {
  union { float f; unsigned u; } a; a.f = f;
  unsigned r = a.u + 0x7FFFu + ((a.u >> 16) & 1u);
  union { unsigned short s; __bf16 b; } o; o.s = (unsigned short)(r >> 16);
  return o.b;
}

__device__ __forceinline__ v8f zero8() {
  v8f z;
#pragma unroll
  for (int i = 0; i < 8; i++) z[i] = 0.f;
  return z;
}

// A and B fragments use the same layout: operand stored row-major with K
// contiguous; lane l -> row (l&15), per-lane K slices [8*(l>>4)..+8) and
// [16+8*(l>>4)..+8) -> two aligned 16B LDS reads.
__device__ __forceinline__ v16bf load_frag(const __bf16* base, int lda, int row0,
                                           int k0, int lane) {
  int r = row0 + (lane & 15);
  int h = lane >> 4;
  const __bf16* p = base + r * lda + k0 + 8 * h;
  fragu u;
  u.h[0] = *(const v8bf*)p;
  u.h[1] = *(const v8bf*)(p + 16);
  return u.v;
}

__device__ __forceinline__ v8f wmma_bf(v16bf a, v16bf b, v8f c) {
  return __builtin_amdgcn_wmma_f32_16x16x32_bf16(false, a, false, b, (short)0, c,
                                                 false, false);
}

__device__ __forceinline__ float siluf(float x) { return x / (1.f + __expf(-x)); }

// CDNA5 async global->LDS copy (16B per lane), tracked on ASYNCcnt.
// VDST operand = per-lane LDS byte offset (workgroup-relative, low 32 bits of
// the generic shared pointer), VADDR = 64-bit global address (GV mode).
__device__ __forceinline__ void async_g2l_b128(void* lds_dst, const void* gsrc) {
  unsigned lds_off = (unsigned)(size_t)lds_dst;
  asm volatile("global_load_async_to_lds_b128 %0, %1, off"
               :: "v"(lds_off), "v"(gsrc)
               : "memory");
}
__device__ __forceinline__ void wait_async0() {
  asm volatile("s_wait_asynccnt 0" ::: "memory");
}

// ---------------------------------------------------------------------------
// fp32 -> bf16 weight conversion
// ---------------------------------------------------------------------------
__global__ __launch_bounds__(256) void f2bf_kernel(const float* __restrict__ s,
                                                   __bf16* __restrict__ d, int n) {
  int i = blockIdx.x * 256 + threadIdx.x;
  if (i < n) d[i] = f2b(s[i]);
}

// ---------------------------------------------------------------------------
// Generic GEMM: C[M,N] = act(A[M,K](f32) * W[N,K](bf16)^T + bias)
// block = 256 thr (8 waves), tile 128x64, K-step 32. act: 0=none, 1=tanh
// ---------------------------------------------------------------------------
__global__ __launch_bounds__(256) void gemm_kernel(
    const float* __restrict__ A, const __bf16* __restrict__ W,
    const float* __restrict__ bias, float* __restrict__ C,
    int M, int N, int K, int act) {
  __shared__ __align__(16) __bf16 As[128 * 40];
  __shared__ __align__(16) __bf16 Bs[64 * 40];
  const int tid = threadIdx.x;
  const int lane = tid & 31, wave = tid >> 5;
  const int m0 = blockIdx.y * 128;
  const int n0 = blockIdx.x * 64;
  const int wm = (wave >> 1) * 32;
  const int wn = (wave & 1) * 32;
  const bool btile_full = (n0 + 64) <= N;   // block-uniform

  v8f acc[2][2];
#pragma unroll
  for (int i = 0; i < 2; i++)
#pragma unroll
    for (int j = 0; j < 2; j++) acc[i][j] = zero8();

  for (int k0 = 0; k0 < K; k0 += 32) {
    __syncthreads();
    {  // stage B (64x32 bf16): async DMA straight into LDS when tile is full
      int r = tid >> 2, c = (tid & 3) * 8;
      __bf16* dst = Bs + r * 40 + c;
      const __bf16* src = W + (size_t)(n0 + r) * K + k0 + c;
      if (btile_full) {
        async_g2l_b128(dst, src);
      } else if ((n0 + r) < N) {
        *(v8bf*)dst = *(const v8bf*)src;
      } else {
#pragma unroll
        for (int t = 0; t < 8; t++) dst[t] = f2b(0.f);
      }
    }
    {  // stage A (128x32 fp32 -> bf16), overlaps with the async B copy
      int r = tid >> 1, c = (tid & 1) * 16;
      bool ok = (m0 + r) < M;
      const float* src = A + (size_t)(m0 + r) * K + k0 + c;
      __bf16* dst = As + r * 40 + c;
      if (ok && (k0 + 32) < K) __builtin_prefetch(src + 32, 0, 1);
#pragma unroll
      for (int v = 0; v < 16; v += 4) {
        float4 f = {0.f, 0.f, 0.f, 0.f};
        if (ok) f = *(const float4*)(src + v);
        dst[v + 0] = f2b(f.x); dst[v + 1] = f2b(f.y);
        dst[v + 2] = f2b(f.z); dst[v + 3] = f2b(f.w);
      }
    }
    if (btile_full) wait_async0();  // barrier orders DScnt only, not ASYNCcnt
    __syncthreads();
    v16bf aF0 = load_frag(As, 40, wm, 0, lane);
    v16bf aF1 = load_frag(As, 40, wm + 16, 0, lane);
    v16bf bF0 = load_frag(Bs, 40, wn, 0, lane);
    v16bf bF1 = load_frag(Bs, 40, wn + 16, 0, lane);
    acc[0][0] = wmma_bf(aF0, bF0, acc[0][0]);
    acc[0][1] = wmma_bf(aF0, bF1, acc[0][1]);
    acc[1][0] = wmma_bf(aF1, bF0, acc[1][0]);
    acc[1][1] = wmma_bf(aF1, bF1, acc[1][1]);
  }

  const int half = lane >> 4, nn = lane & 15;
#pragma unroll
  for (int i = 0; i < 2; i++)
#pragma unroll
    for (int j = 0; j < 2; j++) {
      int nb = n0 + wn + j * 16 + nn;
      if (nb >= N) continue;
      int mb = m0 + wm + i * 16 + 8 * half;
#pragma unroll
      for (int r = 0; r < 8; r++) {
        int m = mb + r;
        if (m < M) {
          float v = acc[i][j][r];
          if (bias) v += bias[nb];
          if (act == 1) v = tanhf(v);
          C[(size_t)m * N + nb] = v;
        }
      }
    }
}

// ---------------------------------------------------------------------------
// Mamba elementwise: dt = softplus(raw+dtb); a = dt * (-exp(Alog))
// ---------------------------------------------------------------------------
__global__ __launch_bounds__(256) void e1a_kernel(
    const float* __restrict__ zx, const float* __restrict__ dtb,
    const float* __restrict__ Alog, float* __restrict__ dt, float* __restrict__ ab,
    int n) {
  int idx = blockIdx.x * 256 + threadIdx.x;
  if (idx >= n) return;
  int t = idx >> 3, h = idx & 7;
  float raw = zx[(size_t)t * 1288 + 1280 + h] + dtb[h];
  float d = raw > 20.f ? raw : log1pf(__expf(raw));
  dt[idx] = d;
  ab[idx] = d * (-__expf(Alog[h]));
}

// causal conv(KC=4) + silu; channels <512 -> xh, xdt=xh*dt; >=512 -> Bm,Cm
__global__ __launch_bounds__(256) void e1b_kernel(
    const float* __restrict__ zx, const float* __restrict__ convw,
    const float* __restrict__ convb, const float* __restrict__ dt,
    float* __restrict__ xh, float* __restrict__ xdt, float* __restrict__ bmcm,
    int L, int n) {
  int idx = blockIdx.x * 256 + threadIdx.x;
  if (idx >= n) return;
  int t = idx / 768, c = idx % 768;
  int b = t / L, l = t % L;
  float acc = convb[c];
#pragma unroll
  for (int k = 0; k < 4; k++) {
    int j = l - 3 + k;
    if (j >= 0) acc += convw[c * 4 + k] * zx[((size_t)(b * L + j)) * 1288 + 512 + c];
  }
  float a = siluf(acc);
  if (c < 512) {
    xh[(size_t)t * 512 + c] = a;
    xdt[(size_t)t * 512 + c] = a * dt[t * 8 + (c >> 6)];
  } else {
    bmcm[(size_t)t * 256 + (c - 512)] = a;
  }
}

// ---------------------------------------------------------------------------
// SSD intra-chunk kernel (block per (b,c)). WMMA: CB, y_intra, states.
// ---------------------------------------------------------------------------
__global__ __launch_bounds__(256) void ssd_chunk_kernel(
    const float* __restrict__ xdt, const float* __restrict__ bmcm,
    const float* __restrict__ ab, float* __restrict__ ybuf,
    float* __restrict__ states, float* __restrict__ s_out,
    float* __restrict__ stot_out, int L, int nc) {
  __shared__ __align__(16) unsigned char smem[56352];
  __bf16* RA  = (__bf16*)smem;             // 9216 bf16: Cm / (xdtT+Mh)
  __bf16* RB  = (__bf16*)(smem + 18432);   // Bm 64x136
  float*  CBf = (float*)(smem + 35840);    // CB 64x68 f32  (union w/ BmT)
  __bf16* BmT = (__bf16*)(smem + 35840);   // BmT 128x72 bf16
  float*  sS  = (float*)(smem + 54272);    // s 64x8
  float*  StS = sS + 512;                  // Stot 8

  const int tid = threadIdx.x;
  const int lane = tid & 31, wave = tid >> 5;
  const int bc = blockIdx.x;
  const int b = bc / nc, c = bc % nc;
  const int q0 = c * 64;
  const int Q = (L - q0 < 64) ? (L - q0) : 64;
  const size_t t0 = (size_t)b * L + q0;

  // stage Cm (RA) and Bm (RB), zero-padded past Q
  for (int it = tid; it < 64 * 32; it += 256) {
    int q = it >> 5, c4 = (it & 31) * 4;
    float4 fb = {0.f, 0.f, 0.f, 0.f}, fc = {0.f, 0.f, 0.f, 0.f};
    if (q < Q) {
      const float* row = bmcm + (t0 + q) * 256;
      fb = *(const float4*)(row + c4);
      fc = *(const float4*)(row + 128 + c4);
    }
    __bf16* pb = RB + q * 136 + c4;
    pb[0] = f2b(fb.x); pb[1] = f2b(fb.y); pb[2] = f2b(fb.z); pb[3] = f2b(fb.w);
    __bf16* pc = RA + q * 136 + c4;
    pc[0] = f2b(fc.x); pc[1] = f2b(fc.y); pc[2] = f2b(fc.z); pc[3] = f2b(fc.w);
  }
  // cumsum s over chunk (one thread per head)
  if (tid < 8) {
    float acc = 0.f;
    for (int q = 0; q < 64; q++) {
      if (q < Q) acc += ab[(t0 + q) * 8 + tid];
      sS[q * 8 + tid] = acc;
      if (q < Q) s_out[(t0 + q) * 8 + tid] = acc;
    }
    StS[tid] = acc;
    stot_out[(size_t)bc * 8 + tid] = acc;
  }
  __syncthreads();

  // CB = Cm (64x128) x Bm^T : 16 tiles, K=128
#pragma unroll
  for (int u = 0; u < 2; u++) {
    int tileId = wave * 2 + u;
    int qt = (tileId >> 2) * 16, kt = (tileId & 3) * 16;
    v8f acc = zero8();
    for (int k0 = 0; k0 < 128; k0 += 32) {
      v16bf aF = load_frag(RA, 136, qt, k0, lane);
      v16bf bF = load_frag(RB, 136, kt, k0, lane);
      acc = wmma_bf(aF, bF, acc);
    }
    int half = lane >> 4, nn = lane & 15;
#pragma unroll
    for (int r = 0; r < 8; r++) CBf[(qt + r + 8 * half) * 68 + kt + nn] = acc[r];
  }
  __syncthreads();

  // per-head y_intra = (CB .* Lmat_h) @ xdt_h
  __bf16* XT = RA;         // 64x72 (p,k)
  __bf16* MH = RA + 4608;  // 64x72 (q,k)
  for (int h = 0; h < 8; h++) {
    for (int it = tid; it < 4096; it += 256) {
      int k = it >> 6, p = it & 63;
      float v = (k < Q) ? xdt[(t0 + k) * 512 + h * 64 + p] : 0.f;
      XT[p * 72 + k] = f2b(v);
    }
    for (int it = tid; it < 4096; it += 256) {
      int q = it >> 6, k = it & 63;
      float v = 0.f;
      if (k <= q) v = CBf[q * 68 + k] * __expf(sS[q * 8 + h] - sS[k * 8 + h]);
      MH[q * 72 + k] = f2b(v);
    }
    __syncthreads();
#pragma unroll
    for (int u = 0; u < 2; u++) {
      int tileId = wave * 2 + u;
      int qt = (tileId >> 2) * 16, pt = (tileId & 3) * 16;
      v8f acc = zero8();
      for (int k0 = 0; k0 < 64; k0 += 32) {
        v16bf aF = load_frag(MH, 72, qt, k0, lane);
        v16bf bF = load_frag(XT, 72, pt, k0, lane);
        acc = wmma_bf(aF, bF, acc);
      }
      int half = lane >> 4, nn = lane & 15;
#pragma unroll
      for (int r = 0; r < 8; r++) {
        int q = qt + r + 8 * half, p = pt + nn;
        if (q < Q) ybuf[(t0 + q) * 512 + h * 64 + p] = acc[r];
      }
    }
    __syncthreads();
  }

  if (nc > 1) {
    // transpose Bm -> BmT (overwrites CB, now dead)
    for (int it = tid; it < 8192; it += 256) {
      int n = it >> 6, k = it & 63;
      BmT[n * 72 + k] = RB[k * 136 + n];
    }
    __syncthreads();
    for (int h = 0; h < 8; h++) {
      for (int it = tid; it < 4096; it += 256) {
        int k = it >> 6, p = it & 63;
        float v = 0.f;
        if (k < Q)
          v = xdt[(t0 + k) * 512 + h * 64 + p] * __expf(StS[h] - sS[k * 8 + h]);
        XT[p * 72 + k] = f2b(v);
      }
      __syncthreads();
#pragma unroll
      for (int u = 0; u < 4; u++) {
        int tileId = wave * 4 + u;
        int pt = (tileId >> 3) * 16, nt = (tileId & 7) * 16;
        v8f acc = zero8();
        for (int k0 = 0; k0 < 64; k0 += 32) {
          v16bf aF = load_frag(XT, 72, pt, k0, lane);
          v16bf bF = load_frag(BmT, 72, nt, k0, lane);
          acc = wmma_bf(aF, bF, acc);
        }
        int half = lane >> 4, nn = lane & 15;
#pragma unroll
        for (int r = 0; r < 8; r++) {
          int p = pt + r + 8 * half, n = nt + nn;
          states[(((size_t)bc * 8 + h) * 64 + p) * 128 + n] = acc[r];
        }
      }
      __syncthreads();
    }
  }
}

// sequential inter-chunk scan; overwrites states with hprev in place
__global__ __launch_bounds__(256) void ssd_scan_kernel(
    float* __restrict__ states, const float* __restrict__ stot, int nc) {
  int bh = blockIdx.x;
  int b = bh >> 3, h = bh & 7;
  float hreg[32];
#pragma unroll
  for (int j = 0; j < 32; j++) hreg[j] = 0.f;
  for (int c = 0; c < nc; c++) {
    size_t base = (((size_t)(b * nc + c) * 8 + h) * 8192);
    float ef = __expf(stot[(size_t)(b * nc + c) * 8 + h]);
#pragma unroll
    for (int j = 0; j < 32; j++) {
      size_t o = base + threadIdx.x + j * 256;
      float st = states[o];
      states[o] = hreg[j];
      hreg[j] = ef * hreg[j] + st;
    }
  }
}

// y += exp(s)*Cm@hprev^T (WMMA) + xh*D; then y*=silu(z); RMSNorm*normw
__global__ __launch_bounds__(256) void ssd_final_kernel(
    float* __restrict__ ybuf, const float* __restrict__ xh,
    const float* __restrict__ zx, const float* __restrict__ bmcm,
    const float* __restrict__ states, const float* __restrict__ s_ws,
    const float* __restrict__ Dv, const float* __restrict__ normw,
    int L, int nc) {
  __shared__ __align__(16) unsigned char smem[36864];
  __bf16* CmS = (__bf16*)smem;             // 64x136
  __bf16* Hb  = (__bf16*)(smem + 17408);   // 64x136 (p,n)
  float*  sL  = (float*)(smem + 34816);    // 64x8

  const int tid = threadIdx.x;
  const int lane = tid & 31, wave = tid >> 5;
  const int bc = blockIdx.x;
  const int b = bc / nc, c = bc % nc;
  const int q0 = c * 64;
  const int Q = (L - q0 < 64) ? (L - q0) : 64;
  const size_t t0 = (size_t)b * L + q0;
  const bool inter = (nc > 1);

  if (inter) {
    for (int it = tid; it < 64 * 32; it += 256) {
      int q = it >> 5, c4 = (it & 31) * 4;
      float4 fc = {0.f, 0.f, 0.f, 0.f};
      if (q < Q) fc = *(const float4*)(bmcm + (t0 + q) * 256 + 128 + c4);
      __bf16* pc = CmS + q * 136 + c4;
      pc[0] = f2b(fc.x); pc[1] = f2b(fc.y); pc[2] = f2b(fc.z); pc[3] = f2b(fc.w);
    }
    for (int it = tid; it < 512; it += 256) {
      int q = it >> 3, h = it & 7;
      sL[it] = (q < Q) ? s_ws[(t0 + q) * 8 + h] : 0.f;
    }
    __syncthreads();
    for (int h = 0; h < 8; h++) {
      size_t base = ((size_t)bc * 8 + h) * 8192;
      for (int it = tid; it < 8192; it += 256) {
        int p = it >> 7, n = it & 127;
        Hb[p * 136 + n] = f2b(states[base + (size_t)p * 128 + n]);
      }
      __syncthreads();
#pragma unroll
      for (int u = 0; u < 2; u++) {
        int tileId = wave * 2 + u;
        int qt = (tileId >> 2) * 16, pt = (tileId & 3) * 16;
        v8f acc = zero8();
        for (int k0 = 0; k0 < 128; k0 += 32) {
          v16bf aF = load_frag(CmS, 136, qt, k0, lane);
          v16bf bF = load_frag(Hb, 136, pt, k0, lane);
          acc = wmma_bf(aF, bF, acc);
        }
        int half = lane >> 4, nn = lane & 15;
#pragma unroll
        for (int r = 0; r < 8; r++) {
          int q = qt + r + 8 * half, p = pt + nn;
          if (q < Q) {
            size_t o = (t0 + q) * 512 + h * 64 + p;
            ybuf[o] += __expf(sL[q * 8 + h]) * acc[r];
          }
        }
      }
      __syncthreads();
    }
    __threadfence_block();
    __syncthreads();
  }

  // per-token: y += xh*D, gate with silu(z), RMS-norm, write back
  for (int qi = wave; qi < 64; qi += 8) {
    if (qi >= Q) continue;
    size_t t = t0 + qi;
    float vv[16];
    float ss = 0.f;
#pragma unroll
    for (int j = 0; j < 16; j++) {
      int cc = lane + 32 * j;
      float y = ybuf[t * 512 + cc] + xh[t * 512 + cc] * Dv[cc >> 6];
      float z = zx[t * 1288 + cc];
      y *= siluf(z);
      vv[j] = y;
      ss += y * y;
    }
#pragma unroll
    for (int o = 16; o >= 1; o >>= 1) ss += __shfl_xor(ss, o, 32);
    float rs = rsqrtf(ss / 512.f + 1e-5f);
#pragma unroll
    for (int j = 0; j < 16; j++) {
      int cc = lane + 32 * j;
      ybuf[t * 512 + cc] = vv[j] * rs * normw[cc];
    }
  }
}

// ---------------------------------------------------------------------------
// groupnorm over (128,T) per sample; output transposed to token-major rows.
// mode 0: row = s*T+t ; mode 1 (TAC): row = ((s>>1)*T+t)*2 + (s&1)
// ---------------------------------------------------------------------------
__global__ __launch_bounds__(256) void gnorm_kernel(
    const float* __restrict__ in, float* __restrict__ out,
    const float* __restrict__ gamma, const float* __restrict__ beta,
    int T, int mode) {
  const int s = blockIdx.x, tid = threadIdx.x;
  const float* X = in + (size_t)s * 128 * T;
  __shared__ float red[256], red2[256], mv[2];
  const int n = 128 * T;
  float sm = 0.f, sq = 0.f;
  for (int i = tid; i < n; i += 256) { float v = X[i]; sm += v; sq += v * v; }
  red[tid] = sm; red2[tid] = sq;
  __syncthreads();
  for (int st = 128; st > 0; st >>= 1) {
    if (tid < st) { red[tid] += red[tid + st]; red2[tid] += red2[tid + st]; }
    __syncthreads();
  }
  if (tid == 0) {
    float mean = red[0] / n;
    float var = red2[0] / n - mean * mean;
    mv[0] = mean; mv[1] = rsqrtf(var + EPS_GN);
  }
  __syncthreads();
  float mean = mv[0], inv = mv[1];
  for (int i = tid; i < n; i += 256) {
    int ch = i / T, t = i % T;
    float v = (X[i] - mean) * inv * gamma[ch] + beta[ch];
    size_t row = (mode == 0) ? ((size_t)s * T + t)
                             : (((size_t)(s >> 1) * T + t) * 2 + (s & 1));
    out[row * 128 + ch] = v;
  }
}

// ---------------------------------------------------------------------------
// small glue kernels
// ---------------------------------------------------------------------------
__global__ __launch_bounds__(256) void reverse_kernel(
    const float* __restrict__ in, float* __restrict__ out, int L, int n) {
  int idx = blockIdx.x * 256 + threadIdx.x;
  if (idx >= n) return;
  int c = idx & 127, t = idx >> 7;
  int b = t / L, l = t % L;
  out[(size_t)t * 128 + c] = in[((size_t)b * L + (L - 1 - l)) * 128 + c];
}

__global__ __launch_bounds__(256) void concat_kernel(
    const float* __restrict__ fout, const float* __restrict__ bout,
    const float* __restrict__ hs, float* __restrict__ cat, int L, int n) {
  int idx = blockIdx.x * 256 + threadIdx.x;
  if (idx >= n) return;
  int cc = idx & 255, t = idx >> 8;
  int b = t / L, l = t % L;
  if (cc < 128) {
    cat[idx] = fout[(size_t)t * 128 + cc] + hs[(size_t)t * 128 + cc];
  } else {
    int c2 = cc - 128;
    cat[idx] = bout[((size_t)b * L + (L - 1 - l)) * 128 + c2] +
               hs[(size_t)t * 128 + c2];
  }
}

__global__ __launch_bounds__(256) void addres_kernel(
    float* __restrict__ hbuf, const float* __restrict__ proj, int T, int n) {
  int idx = blockIdx.x * 256 + threadIdx.x;
  if (idx >= n) return;
  int t = idx % T, c = (idx / T) % 128, s = idx / (128 * T);
  hbuf[idx] += proj[((size_t)s * T + t) * 128 + c];
}

__global__ __launch_bounds__(256) void copy12_kernel(
    const float* __restrict__ h1, float* __restrict__ h2, int n) {
  int idx = blockIdx.x * 256 + threadIdx.x;
  if (idx >= n) return;
  int nb = idx % 7;
  int rest = idx / 7;
  int f = rest % 128;
  int rest2 = rest / 128;
  int t = rest2 % 512, bc = rest2 / 512;
  h2[idx] = h1[(((size_t)(bc * 7 + nb)) * 128 + f) * 512 + t];
}

__global__ __launch_bounds__(256) void copy21tac_kernel(
    const float* __restrict__ h2, float* __restrict__ htac, int n) {
  int idx = blockIdx.x * 256 + threadIdx.x;
  if (idx >= n) return;
  int t = idx % 512, f = (idx / 512) % 128, s = idx / (512 * 128);
  int ch = s & 1, band = (s >> 1) % 7, b = s / 14;
  htac[idx] = h2[((((size_t)(b * 2 + ch)) * 512 + t) * 128 + f) * 7 + band];
}

__global__ __launch_bounds__(256) void mean2_kernel(
    const float* __restrict__ go, float* __restrict__ gm, int n) {
  int idx = blockIdx.x * 256 + threadIdx.x;
  if (idx >= n) return;
  int r2 = idx / 384, c = idx % 384;
  gm[idx] = 0.5f * (go[(size_t)(2 * r2) * 384 + c] +
                    go[(size_t)(2 * r2 + 1) * 384 + c]);
}

__global__ __launch_bounds__(256) void cat768_kernel(
    const float* __restrict__ go, const float* __restrict__ gm,
    float* __restrict__ cat, int n) {
  int idx = blockIdx.x * 256 + threadIdx.x;
  if (idx >= n) return;
  int r = idx / 768, c = idx % 768;
  cat[idx] = (c < 384) ? go[(size_t)r * 384 + c]
                       : gm[(size_t)(r >> 1) * 384 + (c - 384)];
}

__global__ __launch_bounds__(256) void addtac_kernel(
    float* __restrict__ htac, const float* __restrict__ tout, int n) {
  int idx = blockIdx.x * 256 + threadIdx.x;
  if (idx >= n) return;
  int t = idx % 512, f = (idx / 512) % 128, s = idx / (512 * 128);
  size_t r = ((size_t)(s >> 1) * 512 + t) * 2 + (s & 1);
  htac[idx] += tout[r * 128 + f];
}

__global__ __launch_bounds__(256) void copyout_kernel(
    const float* __restrict__ htac, float* __restrict__ out, int n) {
  int idx = blockIdx.x * 256 + threadIdx.x;
  if (idx >= n) return;
  int t = idx % 512;
  int nf = (idx / 512) % 896;
  int ch = (idx / (512 * 896)) % 2;
  int b = idx / (512 * 896 * 2);
  int band = nf >> 7, f = nf & 127;
  out[idx] = htac[(((size_t)((b * 7 + band) * 2 + ch)) * 128 + f) * 512 + t];
}

// ---------------------------------------------------------------------------
// host orchestration
// ---------------------------------------------------------------------------
struct Bufs {
  float *h1, *hs, *hrev, *fout, *bout, *cat2, *proj, *h2, *htac;
  float *zx, *dt, *ab, *xh, *xdt, *bmcm, *ybuf, *sws, *stot, *states;
  __bf16 *winb, *woutb, *projwb, *w1b, *w2b, *w3b;
  const float *convw, *convb, *dtbias, *Alog, *Dh, *normw;
};

static inline int cdiv(int a, int b) { return (a + b - 1) / b; }

static void run_mamba2(hipStream_t st, Bufs& p, const float* xin, float* yout,
                       int pi, int bM, int L) {
  int nc = (L + 63) / 64;
  dim3 g1(cdiv(1288, 64), cdiv(TOKS, 128));
  gemm_kernel<<<g1, 256, 0, st>>>(xin, p.winb + (size_t)pi * 1288 * 128, nullptr,
                                  p.zx, TOKS, 1288, 128, 0);
  e1a_kernel<<<cdiv(TOKS * 8, 256), 256, 0, st>>>(
      p.zx, p.dtbias + pi * 8, p.Alog + pi * 8, p.dt, p.ab, TOKS * 8);
  e1b_kernel<<<cdiv(TOKS * 768, 256), 256, 0, st>>>(
      p.zx, p.convw + (size_t)pi * 768 * 4, p.convb + pi * 768, p.dt, p.xh,
      p.xdt, p.bmcm, L, TOKS * 768);
  ssd_chunk_kernel<<<bM * nc, 256, 0, st>>>(p.xdt, p.bmcm, p.ab, p.ybuf,
                                            p.states, p.sws, p.stot, L, nc);
  if (nc > 1)
    ssd_scan_kernel<<<bM * 8, 256, 0, st>>>(p.states, p.stot, nc);
  ssd_final_kernel<<<bM * nc, 256, 0, st>>>(p.ybuf, p.xh, p.zx, p.bmcm, p.states,
                                            p.sws, p.Dh + pi * 8,
                                            p.normw + pi * 512, L, nc);
  dim3 g2(cdiv(128, 64), cdiv(TOKS, 128));
  gemm_kernel<<<g2, 256, 0, st>>>(p.ybuf, p.woutb + (size_t)pi * 128 * 512,
                                  nullptr, yout, TOKS, 128, 512, 0);
}

static void run_resmamba(hipStream_t st, Bufs& p, float* hbuf, int S, int T,
                         int j, const float* rgamma, const float* rbeta,
                         const float* projbias) {
  gnorm_kernel<<<S, 256, 0, st>>>(hbuf, p.hs, rgamma + j * 128, rbeta + j * 128,
                                  T, 0);
  run_mamba2(st, p, p.hs, p.fout, 2 * j, S, T);
  reverse_kernel<<<cdiv(TOKS * 128, 256), 256, 0, st>>>(p.hs, p.hrev, T,
                                                        TOKS * 128);
  run_mamba2(st, p, p.hrev, p.bout, 2 * j + 1, S, T);
  concat_kernel<<<cdiv(TOKS * 256, 256), 256, 0, st>>>(p.fout, p.bout, p.hs,
                                                       p.cat2, T, TOKS * 256);
  dim3 gp(cdiv(128, 64), cdiv(TOKS, 128));
  gemm_kernel<<<gp, 256, 0, st>>>(p.cat2, p.projwb + (size_t)j * 128 * 256,
                                  projbias + j * 128, p.proj, TOKS, 128, 256, 0);
  addres_kernel<<<cdiv(TOKS * 128, 256), 256, 0, st>>>(hbuf, p.proj, T,
                                                       TOKS * 128);
}

extern "C" void kernel_launch(void* const* d_in, const int* in_sizes, int n_in,
                              void* d_out, int out_size, void* d_ws,
                              size_t ws_size, hipStream_t stream) {
  (void)in_sizes; (void)n_in; (void)out_size; (void)ws_size;
  const float* x      = (const float*)d_in[0];
  const float* Win    = (const float*)d_in[1];
  const float* convw  = (const float*)d_in[2];
  const float* convb  = (const float*)d_in[3];
  const float* dtbias = (const float*)d_in[4];
  const float* Alog   = (const float*)d_in[5];
  const float* Dh     = (const float*)d_in[6];
  const float* normw  = (const float*)d_in[7];
  const float* Wout   = (const float*)d_in[8];
  const float* rgamma = (const float*)d_in[9];
  const float* rbeta  = (const float*)d_in[10];
  const float* projW  = (const float*)d_in[11];
  const float* projb  = (const float*)d_in[12];
  const float* tgamma = (const float*)d_in[13];
  const float* tbeta  = (const float*)d_in[14];
  const float* tW1    = (const float*)d_in[15];
  const float* tb1    = (const float*)d_in[16];
  const float* tW2    = (const float*)d_in[17];
  const float* tb2    = (const float*)d_in[18];
  const float* tW3    = (const float*)d_in[19];
  const float* tb3    = (const float*)d_in[20];

  // workspace carve-out (~305 MB total)
  char* ws = (char*)d_ws;
  size_t off = 0;
  auto alloc = [&](size_t bytes) -> char* {
    char* pp = ws + off;
    off += (bytes + 255) & ~(size_t)255;
    return pp;
  };
  Bufs p;
  p.h1    = (float*)alloc((size_t)TOKS * 128 * 4);
  p.hs    = (float*)alloc((size_t)TOKS * 128 * 4);
  p.hrev  = (float*)alloc((size_t)TOKS * 128 * 4);
  p.fout  = (float*)alloc((size_t)TOKS * 128 * 4);
  p.bout  = (float*)alloc((size_t)TOKS * 128 * 4);
  p.cat2  = (float*)alloc((size_t)TOKS * 256 * 4);
  p.proj  = (float*)alloc((size_t)TOKS * 128 * 4);
  p.h2    = (float*)alloc((size_t)TOKS * 128 * 4);
  p.htac  = (float*)alloc((size_t)TOKS * 128 * 4);
  p.zx    = (float*)alloc((size_t)TOKS * 1288 * 4);
  p.dt    = (float*)alloc((size_t)TOKS * 8 * 4);
  p.ab    = (float*)alloc((size_t)TOKS * 8 * 4);
  p.xh    = (float*)alloc((size_t)TOKS * 512 * 4);
  p.xdt   = (float*)alloc((size_t)TOKS * 512 * 4);
  p.bmcm  = (float*)alloc((size_t)TOKS * 256 * 4);
  p.ybuf  = (float*)alloc((size_t)TOKS * 512 * 4);
  p.sws   = (float*)alloc((size_t)TOKS * 8 * 4);
  p.stot  = (float*)alloc((size_t)2048 * 8 * 4);
  p.states= (float*)alloc((size_t)224 * 8 * 64 * 128 * 4);  // stage-1 only
  p.winb  = (__bf16*)alloc((size_t)4 * 1288 * 128 * 2);
  p.woutb = (__bf16*)alloc((size_t)4 * 128 * 512 * 2);
  p.projwb= (__bf16*)alloc((size_t)2 * 128 * 256 * 2);
  p.w1b   = (__bf16*)alloc((size_t)384 * 128 * 2);
  p.w2b   = (__bf16*)alloc((size_t)384 * 384 * 2);
  p.w3b   = (__bf16*)alloc((size_t)128 * 768 * 2);
  p.convw = convw; p.convb = convb; p.dtbias = dtbias;
  p.Alog = Alog; p.Dh = Dh; p.normw = normw;

  // weight conversions (fp32 -> bf16)
  f2bf_kernel<<<cdiv(4 * 1288 * 128, 256), 256, 0, stream>>>(Win, p.winb,
                                                             4 * 1288 * 128);
  f2bf_kernel<<<cdiv(4 * 128 * 512, 256), 256, 0, stream>>>(Wout, p.woutb,
                                                            4 * 128 * 512);
  f2bf_kernel<<<cdiv(2 * 128 * 256, 256), 256, 0, stream>>>(projW, p.projwb,
                                                            2 * 128 * 256);
  f2bf_kernel<<<cdiv(384 * 128, 256), 256, 0, stream>>>(tW1, p.w1b, 384 * 128);
  f2bf_kernel<<<cdiv(384 * 384, 256), 256, 0, stream>>>(tW2, p.w2b, 384 * 384);
  f2bf_kernel<<<cdiv(128 * 768, 256), 256, 0, stream>>>(tW3, p.w3b, 128 * 768);

  // stage 1: along-time mamba (28 sequences of length 512)
  hipMemcpyAsync(p.h1, x, (size_t)TOKS * 128 * 4, hipMemcpyDeviceToDevice,
                 stream);
  run_resmamba(stream, p, p.h1, 28, 512, 0, rgamma, rbeta, projb);

  // stage 2: across-band mamba (2048 sequences of length 7)
  copy12_kernel<<<cdiv(TOKS * 128, 256), 256, 0, stream>>>(p.h1, p.h2,
                                                           TOKS * 128);
  run_resmamba(stream, p, p.h2, 2048, 7, 1, rgamma, rbeta, projb);

  // TAC
  copy21tac_kernel<<<cdiv(TOKS * 128, 256), 256, 0, stream>>>(p.h2, p.htac,
                                                              TOKS * 128);
  gnorm_kernel<<<28, 256, 0, stream>>>(p.htac, p.hs, tgamma, tbeta, 512, 1);
  {
    dim3 g(cdiv(384, 64), cdiv(TOKS, 128));
    gemm_kernel<<<g, 256, 0, stream>>>(p.hs, p.w1b, tb1, p.xh, TOKS, 384, 128, 1);
  }
  mean2_kernel<<<cdiv(7168 * 384, 256), 256, 0, stream>>>(p.xh, p.xdt,
                                                          7168 * 384);
  {
    dim3 g(cdiv(384, 64), cdiv(7168, 128));
    gemm_kernel<<<g, 256, 0, stream>>>(p.xdt, p.w2b, tb2, p.bmcm, 7168, 384, 384,
                                       1);
  }
  cat768_kernel<<<cdiv(TOKS * 768, 256), 256, 0, stream>>>(p.xh, p.bmcm, p.zx,
                                                           TOKS * 768);
  {
    dim3 g(cdiv(128, 64), cdiv(TOKS, 128));
    gemm_kernel<<<g, 256, 0, stream>>>(p.zx, p.w3b, tb3, p.fout, TOKS, 128, 768,
                                       1);
  }
  addtac_kernel<<<cdiv(TOKS * 128, 256), 256, 0, stream>>>(p.htac, p.fout,
                                                           TOKS * 128);

  copyout_kernel<<<cdiv(TOKS * 128, 256), 256, 0, stream>>>(p.htac,
                                                            (float*)d_out,
                                                            TOKS * 128);
}
